// Conv_58067957842195
// MI455X (gfx1250) — compile-verified, matrix operand order checked by reading, not run on hardware
//
#include <hip/hip_runtime.h>

// ---------------------------------------------------------------------------
// Modulated 3x3 conv (StyleGAN2-style) for MI455X / gfx1250.
// Implicit GEMM per batch: out(o=128, n=16384) = Wmod_b(128,1152) x X_b(1152,16384)
// bf16 WMMA (f32 accumulate), wmod staged in d_ws as [b][tap][o][i] bf16.
// ---------------------------------------------------------------------------

typedef __attribute__((ext_vector_type(16))) __bf16 v16bf;
typedef __attribute__((ext_vector_type(8)))  float  v8f;

#define B_   16
#define C_   128
#define H_   128
#define W_   128
#define KK   9            // 3x3 taps
#define KDIM (C_ * KK)    // 1152
#define TN   64           // N-tile (pixels) per block
#define XCOLS 66          // 64 + 2 halo columns
#define XROW  136         // padded LDS row length (bf16 elems): 128 + 8 pad, keeps 16B align

__device__ __forceinline__ unsigned short f32_to_bf16(float f) {
  // round-to-nearest-even
  unsigned int u = __float_as_uint(f);
  unsigned int r = u + 0x7FFFu + ((u >> 16) & 1u);
  return (unsigned short)(r >> 16);
}

// ---------------------------------------------------------------------------
// Kernel 1: style/weight normalization + modulation + demodulation.
// One block per (o, b). Writes wmod[b][tap][o][i] as bf16 (i-contiguous).
// ---------------------------------------------------------------------------
__global__ __launch_bounds__(256) void modulate_kernel(
    const float* __restrict__ style,      // (B, C)
    const float* __restrict__ weight,     // (C, C, 3, 3)
    unsigned short* __restrict__ wmod)    // (B, 9, C, C) bf16
{
  const int o   = blockIdx.x;
  const int b   = blockIdx.y;
  const int tid = threadIdx.x;

  __shared__ float red[256];

  // --- smax = max_i |style[b,i]| ---
  float m = 0.0f;
  if (tid < C_) m = fabsf(style[b * C_ + tid]);
  red[tid] = m;
  __syncthreads();
#pragma unroll
  for (int s = 128; s > 0; s >>= 1) {
    if (tid < s) red[tid] = fmaxf(red[tid], red[tid + s]);
    __syncthreads();
  }
  const float smax = red[0];
  __syncthreads();

  // --- wmax = max_e |weight[o, e]| over e in [0,1152) ---
  float wm = 0.0f;
  for (int e = tid; e < KDIM; e += 256)
    wm = fmaxf(wm, fabsf(weight[(size_t)o * KDIM + e]));
  red[tid] = wm;
  __syncthreads();
#pragma unroll
  for (int s = 128; s > 0; s >>= 1) {
    if (tid < s) red[tid] = fmaxf(red[tid], red[tid + s]);
    __syncthreads();
  }
  const float wmax = red[0];
  __syncthreads();

  const float inv_s  = 1.0f / smax;
  const float wscale = rsqrtf((float)KDIM) / wmax;   // 1/sqrt(1152) / wmax

  // --- sum of squares of modulated weights for this (b,o) ---
  float ss = 0.0f;
  for (int e = tid; e < KDIM; e += 256) {
    const int i = e / KK;
    const float t = weight[(size_t)o * KDIM + e] * wscale *
                    (style[b * C_ + i] * inv_s);
    ss += t * t;
  }
  red[tid] = ss;
  __syncthreads();
#pragma unroll
  for (int s = 128; s > 0; s >>= 1) {
    if (tid < s) red[tid] += red[tid + s];
    __syncthreads();
  }
  const float coe = rsqrtf(red[0] + 1e-8f);

  // --- write demodulated weights, reorganized [b][tap][o][i], bf16 ---
  for (int e = tid; e < KDIM; e += 256) {
    const int i   = e / KK;
    const int tap = e % KK;
    const float t = weight[(size_t)o * KDIM + e] * wscale *
                    (style[b * C_ + i] * inv_s) * coe;
    wmod[(((size_t)b * KK + tap) * C_ + o) * C_ + i] = f32_to_bf16(t);
  }
}

// ---------------------------------------------------------------------------
// Kernel 2: implicit-GEMM conv via v_wmma_f32_16x16x32_bf16.
// Block = (b, h, 64-pixel tile), 256 threads = 8 wave32.
// Wave w computes output channels [16w, 16w+16) x 64 pixels (4 WMMA N-tiles).
// ---------------------------------------------------------------------------
__global__ __launch_bounds__(256) void conv_kernel(
    const float* __restrict__ x,              // (B, C, H, W) f32
    const unsigned short* __restrict__ wmod,  // (B, 9, C, C) bf16
    float* __restrict__ out)                  // (B, C, H, W) f32
{
  __shared__ unsigned short xs[3 * XCOLS * XROW];   // [row][col][chan] bf16

  const int n0  = blockIdx.x * TN;
  const int h   = blockIdx.y;
  const int b   = blockIdx.z;
  const int tid = threadIdx.x;

  // ---- stage input halo: 3 rows x 66 cols x 128 ch, f32 -> bf16, transposed
  // 3*128*66 = 25344 = 99 * 256 : every thread does exactly 99 iterations.
  for (int e = tid; e < 3 * C_ * XCOLS; e += 256) {
    const int c = e % XCOLS;          // consecutive tid -> consecutive col (coalesced)
    const int p = e / XCOLS;
    const int i = p % C_;
    const int r = p / C_;
    const int hh = h - 1 + r;
    const int ww = n0 - 1 + c;
    float v = 0.0f;
    if ((unsigned)hh < (unsigned)H_ && (unsigned)ww < (unsigned)W_)
      v = x[(((size_t)b * C_ + i) * H_ + hh) * W_ + ww];
    xs[(r * XCOLS + c) * XROW + i] = f32_to_bf16(v);
  }
  __syncthreads();

  const int lane   = tid & 31;          // wave32
  const int wave   = tid >> 5;          // 0..7
  const int ml     = lane & 15;
  const int khalf  = (lane >> 4) << 3;  // 0 or 8 : K half per ISA 16-bit layout
  const int o_base = wave * 16;

  v8f acc[4];
#pragma unroll
  for (int t = 0; t < 4; ++t)
#pragma unroll
    for (int r = 0; r < 8; ++r) acc[t][r] = 0.0f;

  union Frag { uint4 u[2]; v16bf v; };

#pragma unroll
  for (int tap = 0; tap < KK; ++tap) {
    const int dh = tap / 3;
    const int dw = tap % 3;
    // A tile for this (b, tap): wmod[b][tap][o][i], row for this lane's M
    const unsigned short* Atap =
        wmod + (((size_t)b * KK + tap) * C_ + (o_base + ml)) * C_;
    if (tap + 1 < KK)
      __builtin_prefetch(Atap + C_ * C_, 0, 1);  // next tap's A (L2-resident)

#pragma unroll
    for (int kc = 0; kc < C_; kc += 32) {
      Frag A;
      A.u[0] = *(const uint4*)(Atap + kc + khalf);        // K = kc+khalf .. +7
      A.u[1] = *(const uint4*)(Atap + kc + khalf + 16);   // K = kc+khalf+16 .. +23

#pragma unroll
      for (int nt = 0; nt < 4; ++nt) {
        const int c = nt * 16 + ml + dw;                  // halo col index, 0..65
        const unsigned short* Bp = &xs[(dh * XCOLS + c) * XROW + kc + khalf];
        Frag Bf;
        Bf.u[0] = *(const uint4*)(Bp);                    // ds_load_b128
        Bf.u[1] = *(const uint4*)(Bp + 16);
        acc[nt] = __builtin_amdgcn_wmma_f32_16x16x32_bf16(
            false, A.v, false, Bf.v, (short)0, acc[nt], false, false);
      }
    }
  }

  // ---- store: C/D layout -> VGPR e: lanes 0-15 M=e, lanes 16-31 M=e+8
  const int ohalf = (lane >> 4) << 3;
#pragma unroll
  for (int nt = 0; nt < 4; ++nt) {
    const int n = n0 + nt * 16 + ml;
#pragma unroll
    for (int r = 0; r < 8; ++r) {
      const int o = o_base + r + ohalf;
      out[(((size_t)b * C_ + o) * H_ + h) * W_ + n] = acc[nt][r];
    }
  }
}

// ---------------------------------------------------------------------------
extern "C" void kernel_launch(void* const* d_in, const int* in_sizes, int n_in,
                              void* d_out, int out_size, void* d_ws, size_t ws_size,
                              hipStream_t stream) {
  (void)in_sizes; (void)n_in; (void)out_size; (void)ws_size;
  const float* x      = (const float*)d_in[0];   // (16,128,128,128)
  const float* style  = (const float*)d_in[1];   // (16,128)
  const float* weight = (const float*)d_in[2];   // (128,128,3,3)
  float* out          = (float*)d_out;
  unsigned short* wmod = (unsigned short*)d_ws;  // 16*9*128*128 bf16 = 4.5 MB

  modulate_kernel<<<dim3(C_, B_), 256, 0, stream>>>(style, weight, wmod);
  conv_kernel<<<dim3(W_ / TN, H_, B_), 256, 0, stream>>>(x, wmod, out);
}
